// FeatureRelationDecoderV3_72834055406378
// MI455X (gfx1250) — compile-verified, matrix operand order checked by reading, not run on hardware
//
#include <hip/hip_runtime.h>

typedef __attribute__((ext_vector_type(16))) _Float16 v16h;
typedef __attribute__((ext_vector_type(8)))  float    v8f;
typedef __attribute__((ext_vector_type(4)))  float    v4f;

#define R_REL 32  // relations per pair (reference R)

// ---------------------------------------------------------------------------
// Kernel A: ns[i] = 1 iff row i of seg_matrix has any nonzero entry
// (segment size > 1). Row-sum done with V_WMMA_F32_16X16X32_F16:
//   A = 16x32 tile of seg (0/1 values, exact in f16), B = all-ones 32x16,
//   C accumulates f32 row sums (<=191, exact). One wave per 16 rows.
// ---------------------------------------------------------------------------
__global__ __launch_bounds__(256) void seg_rowsum_wmma(const float* __restrict__ seg,
                                                       int* __restrict__ ns,
                                                       int n) {
  const int wave = threadIdx.x >> 5;
  const int lane = threadIdx.x & 31;
  const int m0   = (blockIdx.x * 8 + wave) * 16;   // first row of this wave's tile
  const int m    = m0 + (lane & 15);               // A-matrix row held by this lane
  const int kb   = (lane < 16) ? 0 : 8;            // K base per ISA A layout

  v16h b;
#pragma unroll
  for (int h = 0; h < 16; ++h) b[h] = (_Float16)1.0f;

  v8f c = {};
  const float* row = seg + (size_t)m * n;

  for (int k0 = 0; k0 < n; k0 += 32) {
    // lanes 0-15: halves 0..7 -> K=k0+0..7,  halves 8..15 -> K=k0+16..23
    // lanes16-31: halves 0..7 -> K=k0+8..15, halves 8..15 -> K=k0+24..31
    const v4f x0 = ((const v4f*)(row + k0 + kb))[0];
    const v4f x1 = ((const v4f*)(row + k0 + kb))[1];
    const v4f y0 = ((const v4f*)(row + k0 + kb + 16))[0];
    const v4f y1 = ((const v4f*)(row + k0 + kb + 16))[1];
    v16h a;
    a[0]  = (_Float16)x0.x; a[1]  = (_Float16)x0.y; a[2]  = (_Float16)x0.z; a[3]  = (_Float16)x0.w;
    a[4]  = (_Float16)x1.x; a[5]  = (_Float16)x1.y; a[6]  = (_Float16)x1.z; a[7]  = (_Float16)x1.w;
    a[8]  = (_Float16)y0.x; a[9]  = (_Float16)y0.y; a[10] = (_Float16)y0.z; a[11] = (_Float16)y0.w;
    a[12] = (_Float16)y1.x; a[13] = (_Float16)y1.y; a[14] = (_Float16)y1.z; a[15] = (_Float16)y1.w;
    c = __builtin_amdgcn_wmma_f32_16x16x32_f16(false, a, false, b,
                                               (short)0, c, false, false);
  }

  // D layout: c[g] at lane L<16 is D[M=g][N=L]; at lane L>=16 is D[M=g+8][N=L-16].
  // B is all-ones, so every column holds the row sum.
  if (lane == 0) {
#pragma unroll
    for (int g = 0; g < 8; ++g) ns[m0 + g] = (c[g] > 0.5f) ? 1 : 0;
  } else if (lane == 16) {
#pragma unroll
    for (int g = 0; g < 8; ++g) ns[m0 + 8 + g] = (c[g] > 0.5f) ? 1 : 0;
  }
}

// ---------------------------------------------------------------------------
// Kernel B: stream the [N,N,32] output.
// Block = 256 threads handles row i = blockIdx.y, 256 columns starting at
// blockIdx.x*256. Thread t -> column lane jl = t/8, relation quad = t%8.
// Each wave's 32 lanes cover 4 consecutive j fully -> 512 B contiguous NT
// stores per iteration. 8 iterations cover the 256-column tile.
// ---------------------------------------------------------------------------
__global__ __launch_bounds__(256) void pair_stream(const float* __restrict__ z,
                                                   const float* __restrict__ seg,
                                                   const int* __restrict__ batch,
                                                   const int* __restrict__ ns,
                                                   float* __restrict__ out,
                                                   int n) {
  const int i    = blockIdx.y;
  const int quad = threadIdx.x & 7;    // which float4 of the 32 relations
  const int jl   = threadIdx.x >> 3;   // 0..31
  const int j0   = blockIdx.x * 256;

  const v4f zi  = ((const v4f*)(z + (size_t)i * R_REL))[quad];
  const int bi  = batch[i];
  const int nsi = ns[i];
  v4f base = { (quad == 0) ? 1.0f : 0.0f, 0.0f, 0.0f, 0.0f };

  const float* segrow = seg + (size_t)i * n;
  float* orow = out + (size_t)i * n * R_REL;

#pragma unroll
  for (int it = 0; it < 8; ++it) {
    const int j = j0 + it * 32 + jl;
    if (it < 7) __builtin_prefetch(segrow + j + 32, 0, 0);  // global_prefetch_b8

    const bool cond = (nsi != 0) && (batch[j] == bi) && (ns[j] != 0) &&
                      (segrow[j] == 0.0f) && (j != i);

    const v4f zj = ((const v4f*)(z + (size_t)j * R_REL))[quad];
    v4f v = cond ? (zi * zj) : base;

    // Non-temporal 128-bit store: 302 MB write stream must not evict z/seg.
    __builtin_nontemporal_store(
        v, (v4f*)(orow + (size_t)j * R_REL + quad * 4));
  }
}

// ---------------------------------------------------------------------------
// Inputs (setup_inputs order): 0=z [N,R] f32, 1=seg_matrix [N,N] f32,
// 2=cls_label [N] i32 (unused), 3=batch [N] i32.  Output: [N,N,R] f32.
// ---------------------------------------------------------------------------
extern "C" void kernel_launch(void* const* d_in, const int* in_sizes, int n_in,
                              void* d_out, int out_size, void* d_ws, size_t ws_size,
                              hipStream_t stream) {
  const float* z     = (const float*)d_in[0];
  const float* seg   = (const float*)d_in[1];
  const int*   batch = (const int*)d_in[3];
  float*       out   = (float*)d_out;

  const int n = in_sizes[3];        // N (= batch length) = 1536
  int* ns = (int*)d_ws;             // per-node non-singleton flags (N ints)

  // Stage 1: non-singleton flags via WMMA row reduction (12 blocks x 8 waves).
  seg_rowsum_wmma<<<n / 128, 256, 0, stream>>>(seg, ns, n);

  // Stage 2: stream the 302 MB output.
  dim3 grid(n / 256, n);
  pair_stream<<<grid, 256, 0, stream>>>(z, seg, batch, ns, out, n);
}
